// MLACausalSelfAttention_32762010534334
// MI455X (gfx1250) — matrix operand order repaired and checked by reading, not compiled, not run on hardware
//
#include <hip/hip_runtime.h>
#include <hip/hip_bf16.h>
#include <cstdint>

// ---------------- types ----------------
typedef __bf16    v16bf __attribute__((ext_vector_type(16)));
typedef float     v8f   __attribute__((ext_vector_type(8)));
typedef uint32_t  u32x4 __attribute__((ext_vector_type(4)));
typedef int       i32x4 __attribute__((ext_vector_type(4)));

#define NB    2
#define NT    2048
#define NC    2048
#define NHEAD 16
#define NKV   4
#define HDIM  128

// ---------- CDNA5 async global->LDS path (compile-safe guards) ----------
#if defined(__has_builtin)
#if __has_builtin(__builtin_amdgcn_global_load_async_to_lds_b128)
#define HAVE_ASYNC_LDS 1
#endif
#endif

#ifdef HAVE_ASYNC_LDS
typedef __attribute__((address_space(1))) i32x4* gas_v4i;   // global, typed
typedef __attribute__((address_space(3))) i32x4* las_v4i;   // LDS, typed
__device__ __forceinline__ void async_ld16(const void* g, void* l) {
  // generic->AS1: same bit pattern; generic LDS addr low 32 bits == LDS offset (ISA 10.2)
  __builtin_amdgcn_global_load_async_to_lds_b128(
      (gas_v4i)(unsigned long long)g,
      (las_v4i)(unsigned int)(unsigned long long)l,
      0, 0);
}
__device__ __forceinline__ void wait_async0() {
#if __has_builtin(__builtin_amdgcn_s_wait_asynccnt)
  __builtin_amdgcn_s_wait_asynccnt(0);
#else
  asm volatile("s_wait_asynccnt 0" ::: "memory");
#endif
}
#endif

__device__ __forceinline__ uint16_t f2bf(float f) {
  uint32_t u = __float_as_uint(f);
  uint32_t r = u + 0x7FFFu + ((u >> 16) & 1u);   // round-to-nearest-even
  return (uint16_t)(r >> 16);
}

__device__ __forceinline__ v8f v8f_zero() {
  v8f z = {0.f,0.f,0.f,0.f,0.f,0.f,0.f,0.f};
  return z;
}

// A-matrix fragment (16x32 bf16): lane<16 holds row m=lane, K {0..7,16..23};
// lane>=16 holds row m=lane-16, K {8..15,24..31}.  p = row start + k0.
__device__ __forceinline__ v16bf frag_a(const uint16_t* p, int lane) {
  int kb = (lane & 16) ? 8 : 0;
  union { v16bf v; u32x4 u[2]; } f;
  f.u[0] = *(const u32x4*)(p + kb);
  f.u[1] = *(const u32x4*)(p + 16 + kb);
  return f.v;
}

// B-matrix fragment (32x16 bf16, fed from N-major storage Bt[n][k]):
// lane<16 holds col n=lane, K {0..7,16..23}; lane>=16 holds col n=lane-16, K {8..15,24..31}.
__device__ __forceinline__ v16bf frag_b(const uint16_t* p, int lane) {
  int kb = (lane & 16) ? 8 : 0;
  union { v16bf v; u32x4 u[2]; } f;
  f.u[0] = *(const u32x4*)(p + kb);
  f.u[1] = *(const u32x4*)(p + 16 + kb);
  return f.v;
}

__device__ __forceinline__ v8f wmma_bf16(v16bf a, v16bf b, v8f c) {
  return __builtin_amdgcn_wmma_f32_16x16x32_bf16(false, a, false, b, (short)0, c,
                                                 false, false);
}

// ---------------- elementwise converts ----------------
__global__ void cvt_kernel(const float* __restrict__ in, uint16_t* __restrict__ out, int n) {
  int i = blockIdx.x * 256 + threadIdx.x;
  if (i < n) out[i] = f2bf(in[i]);
}

// in: (K,N) f32 row-major -> out: (N,K) bf16 row-major
__global__ void cvt_tr_kernel(const float* __restrict__ in, uint16_t* __restrict__ out,
                              int K, int N) {
  int i = blockIdx.x * 256 + threadIdx.x;
  if (i < K * N) {
    int k = i / N, n = i % N;
    out[(size_t)n * K + k] = f2bf(in[i]);
  }
}

// ---------------- generic bf16 GEMM: C(MxN) = A(MxK) * Bt(NxK)^T ----------------
// 128x128 tile / WG, BK=64, double-buffered async tiles,
// 8 waves each own 32x64 (2x4 WMMA tiles, 2 k-steps).
template <bool OUT_BF16>
__global__ __launch_bounds__(256) void gemm_bf16_kernel(
    const uint16_t* __restrict__ A, const uint16_t* __restrict__ Bt,
    void* __restrict__ Cout, int M, int N, int K)
{
  __shared__ __align__(16) uint16_t sA[2][128 * 64];
  __shared__ __align__(16) uint16_t sB[2][128 * 64];

  const int tid  = threadIdx.x;
  const int lane = tid & 31;
  const int wave = tid >> 5;
  const int wm   = wave >> 1;   // 0..3 : 32-row band
  const int wn   = wave & 1;    // 0..1 : 64-col band
  const int ln   = lane & 15;
  const int bm   = blockIdx.y * 128;
  const int bn   = blockIdx.x * 128;

  v8f acc[2][4];
#pragma unroll
  for (int i = 0; i < 2; i++)
#pragma unroll
    for (int j = 0; j < 4; j++) acc[i][j] = v8f_zero();

  auto load_tile = [&](int buf, int k0) {
#pragma unroll
    for (int i = 0; i < 4; i++) {
      int l = tid + i * 256;               // 1024 16B chunks per tile
      int r = l >> 3, c = (l & 7) << 3;
#ifdef HAVE_ASYNC_LDS
      async_ld16(&A [(size_t)(bm + r) * K + k0 + c], &sA[buf][r * 64 + c]);
      async_ld16(&Bt[(size_t)(bn + r) * K + k0 + c], &sB[buf][r * 64 + c]);
#else
      *(u32x4*)&sA[buf][r * 64 + c] = *(const u32x4*)&A [(size_t)(bm + r) * K + k0 + c];
      *(u32x4*)&sB[buf][r * 64 + c] = *(const u32x4*)&Bt[(size_t)(bn + r) * K + k0 + c];
#endif
    }
  };

  load_tile(0, 0);
#ifdef HAVE_ASYNC_LDS
  wait_async0();
#endif
  __syncthreads();

  int cur = 0;
  for (int k0 = 0; k0 < K; k0 += 64) {
    if (k0 + 64 < K) load_tile(1 - cur, k0 + 64);   // prefetch next tile (overlaps WMMA)

#pragma unroll
    for (int ks = 0; ks < 2; ks++) {
      v16bf af[2], bfm[4];
#pragma unroll
      for (int mt = 0; mt < 2; mt++)
        af[mt] = frag_a(&sA[cur][(wm * 32 + mt * 16 + ln) * 64 + ks * 32], lane);
#pragma unroll
      for (int nt = 0; nt < 4; nt++)
        bfm[nt] = frag_b(&sB[cur][(wn * 64 + nt * 16 + ln) * 64 + ks * 32], lane);
#pragma unroll
      for (int mt = 0; mt < 2; mt++)
#pragma unroll
        for (int nt = 0; nt < 4; nt++)
          acc[mt][nt] = wmma_bf16(af[mt], bfm[nt], acc[mt][nt]);
    }

#ifdef HAVE_ASYNC_LDS
    wait_async0();        // my prefetch landed
#endif
    __syncthreads();      // everyone done reading cur + everyone's prefetch landed
    cur ^= 1;
  }

  const int hl = lane >> 4;
#pragma unroll
  for (int mt = 0; mt < 2; mt++)
#pragma unroll
    for (int nt = 0; nt < 4; nt++) {
      int col = bn + wn * 64 + nt * 16 + ln;
#pragma unroll
      for (int r = 0; r < 8; r++) {
        int row = bm + wm * 32 + mt * 16 + r + hl * 8;
        float v = acc[mt][nt][r];
        if (OUT_BF16) ((uint16_t*)Cout)[(size_t)row * N + col] = f2bf(v);
        else          ((float*)   Cout)[(size_t)row * N + col] = v;
      }
    }
}

// ---------------- q prep: partial rope + rms, [b,t,h*128] f32 -> [b,h,t,128] bf16 ----
__global__ __launch_bounds__(256) void qprep_kernel(
    const float* __restrict__ q, const float* __restrict__ cosp,
    const float* __restrict__ sinp, uint16_t* __restrict__ qbf)
{
  int wid  = blockIdx.x * 8 + (threadIdx.x >> 5);
  int lane = threadIdx.x & 31;
  int b   = wid / (NT * NHEAD);
  int rem = wid % (NT * NHEAD);
  int h   = rem / NT;
  int t   = rem % NT;
  const float* row = q + ((size_t)(b * NT + t)) * NC + h * HDIM;
  float x1 = row[lane], x2 = row[lane + 32];
  float c = cosp[t * 64 + lane], s = sinp[t * 64 + lane];
  float v0 =  x1 * c + x2 * s;
  float v1 = -x1 * s + x2 * c;
  float v2 = row[lane + 64];
  float v3 = row[lane + 96];
  float ss = v0 * v0 + v1 * v1 + v2 * v2 + v3 * v3;
#pragma unroll
  for (int m = 1; m < 32; m <<= 1) ss += __shfl_xor(ss, m, 32);
  float inv = rsqrtf(ss * (1.0f / 128.0f) + 1.1920929e-7f);
  uint16_t* out = qbf + ((size_t)((b * NHEAD + h) * NT + t)) * HDIM;
  out[lane]      = f2bf(v0 * inv);
  out[lane + 32] = f2bf(v1 * inv);
  out[lane + 64] = f2bf(v2 * inv);
  out[lane + 96] = f2bf(v3 * inv);
}

// ---------------- k prep: full rope on rope-part + concat nope + rms ----------------
__global__ __launch_bounds__(256) void kprep_kernel(
    const float* __restrict__ krope, const float* __restrict__ knope,
    const float* __restrict__ cosp, const float* __restrict__ sinp,
    uint16_t* __restrict__ kbf)
{
  int wid  = blockIdx.x * 8 + (threadIdx.x >> 5);
  int lane = threadIdx.x & 31;
  int b   = wid / (NT * NKV);
  int rem = wid % (NT * NKV);
  int kvh = rem / NT;
  int t   = rem % NT;
  const float* rr = krope + ((size_t)(b * NT + t)) * (NKV * 64) + kvh * 64;
  const float* rn = knope + ((size_t)(b * NT + t)) * (NKV * 64) + kvh * 64;
  float x1 = rr[lane], x2 = rr[lane + 32];
  float c = cosp[t * 64 + lane], s = sinp[t * 64 + lane];
  float v0 =  x1 * c + x2 * s;
  float v1 = -x1 * s + x2 * c;
  float v2 = rn[lane];
  float v3 = rn[lane + 32];
  float ss = v0 * v0 + v1 * v1 + v2 * v2 + v3 * v3;
#pragma unroll
  for (int m = 1; m < 32; m <<= 1) ss += __shfl_xor(ss, m, 32);
  float inv = rsqrtf(ss * (1.0f / 128.0f) + 1.1920929e-7f);
  uint16_t* out = kbf + ((size_t)((b * NKV + kvh) * NT + t)) * HDIM;
  out[lane]      = f2bf(v0 * inv);
  out[lane + 32] = f2bf(v1 * inv);
  out[lane + 64] = f2bf(v2 * inv);
  out[lane + 96] = f2bf(v3 * inv);
}

// ---------------- v transpose: [b,t,kvh*128] f32 -> [b,kvh,128,T] bf16 ----------------
__global__ void vtr_kernel(const float* __restrict__ v, uint16_t* __restrict__ vt) {
  int i   = blockIdx.x * 256 + threadIdx.x;   // over B*T*512
  int b   = i / (NT * 512);
  int rem = i % (NT * 512);
  int t   = rem / 512;
  int c   = rem % 512;
  int kvh = c >> 7, d = c & 127;
  vt[((size_t)((b * NKV + kvh) * HDIM + d)) * NT + t] = f2bf(v[i]);
}

// ---------------- flash attention (wave32 WMMA, double-buffered K/V) ----------------
// grid: (T/128, B*NHEAD); 8 waves, each owns 16 q rows of the 128-row tile.
__global__ __launch_bounds__(256) void fa_kernel(
    const uint16_t* __restrict__ Qb, const uint16_t* __restrict__ Kb,
    const uint16_t* __restrict__ Vt, uint16_t* __restrict__ Yb)
{
  __shared__ __align__(16) uint16_t sK[2][64 * 128];   // keys [key][d]
  __shared__ __align__(16) uint16_t sV[2][128 * 64];   // v^T  [d][key]
  __shared__ __align__(16) uint16_t sP[8][16 * 64];    // per-wave P scratch

  const int tid  = threadIdx.x;
  const int lane = tid & 31;
  const int wave = tid >> 5;
  const int ln   = lane & 15;
  const int hl   = lane >> 4;
  const int qt   = blockIdx.x;
  const int bh   = blockIdx.y;
  const int b = bh >> 4, h = bh & 15, kvh = h >> 2;

  const uint16_t* Qh = Qb + ((size_t)(b * NHEAD + h) * NT) * HDIM;
  const uint16_t* Kh = Kb + ((size_t)(b * NKV + kvh) * NT) * HDIM;
  const uint16_t* Vh = Vt + ((size_t)(b * NKV + kvh) * HDIM) * NT;

  const int qrow0 = qt * 128 + wave * 16;

  v16bf qf[4];
#pragma unroll
  for (int ks = 0; ks < 4; ks++)
    qf[ks] = frag_a(Qh + (size_t)(qrow0 + ln) * HDIM + ks * 32, lane);

  float mrow[8], lrow[8], alpha[8];
  v8f o[8];
#pragma unroll
  for (int r = 0; r < 8; r++) { mrow[r] = -3.0e38f; lrow[r] = 0.f; }
#pragma unroll
  for (int dt = 0; dt < 8; dt++) o[dt] = v8f_zero();

  const int   nblocks = qt * 2 + 2;        // causal: key blocks with k <= qmax
  const float scale   = 0.08838834764831845f;  // 1/sqrt(128)

  auto load_tiles = [&](int buf, int jb) {
#pragma unroll
    for (int i = 0; i < 4; i++) {
      int l  = tid + i * 256;
      int kr = l >> 4, kc = (l & 15) << 3;
      int vr = l >> 3, vc = (l & 7) << 3;
#ifdef HAVE_ASYNC_LDS
      async_ld16(&Kh[(size_t)(jb * 64 + kr) * HDIM + kc], &sK[buf][kr * 128 + kc]);
      async_ld16(&Vh[(size_t)vr * NT + jb * 64 + vc],     &sV[buf][vr * 64 + vc]);
#else
      *(u32x4*)&sK[buf][kr * 128 + kc] =
          *(const u32x4*)&Kh[(size_t)(jb * 64 + kr) * HDIM + kc];
      *(u32x4*)&sV[buf][vr * 64 + vc] =
          *(const u32x4*)&Vh[(size_t)vr * NT + jb * 64 + vc];
#endif
    }
  };

  load_tiles(0, 0);
#ifdef HAVE_ASYNC_LDS
  wait_async0();
#endif
  __syncthreads();

  int cur = 0;
  for (int jb = 0; jb < nblocks; ++jb) {
    if (jb + 1 < nblocks) load_tiles(1 - cur, jb + 1);   // prefetch next key block

    // S = Q K^T  (16 x 64 per wave)
    v8f s[4];
#pragma unroll
    for (int nt = 0; nt < 4; nt++) {
      s[nt] = v8f_zero();
#pragma unroll
      for (int ks = 0; ks < 4; ks++) {
        v16bf kf = frag_b(&sK[cur][(nt * 16 + ln) * 128 + ks * 32], lane);
        s[nt] = wmma_bf16(qf[ks], kf, s[nt]);
      }
    }

    // scale + causal mask
#pragma unroll
    for (int nt = 0; nt < 4; nt++) {
      int ng = jb * 64 + nt * 16 + ln;
#pragma unroll
      for (int r = 0; r < 8; r++) {
        int mg = qrow0 + r + hl * 8;
        float v = s[nt][r] * scale;
        s[nt][r] = (ng > mg) ? -1.0e30f : v;
      }
    }

    // online softmax, per row (rows live across 16 lanes of each half-wave)
#pragma unroll
    for (int r = 0; r < 8; r++) {
      float rm = s[0][r];
#pragma unroll
      for (int nt = 1; nt < 4; nt++) rm = fmaxf(rm, s[nt][r]);
#pragma unroll
      for (int m = 1; m < 16; m <<= 1) rm = fmaxf(rm, __shfl_xor(rm, m, 32));
      float mnew = fmaxf(mrow[r], rm);
      alpha[r] = __expf(mrow[r] - mnew);
      float rs = 0.f;
#pragma unroll
      for (int nt = 0; nt < 4; nt++) {
        float p = __expf(s[nt][r] - mnew);
        rs += p;
        sP[wave][(r + hl * 8) * 64 + nt * 16 + ln] = f2bf(p);
      }
#pragma unroll
      for (int m = 1; m < 16; m <<= 1) rs += __shfl_xor(rs, m, 32);
      lrow[r] = lrow[r] * alpha[r] + rs;
      mrow[r] = mnew;
    }
#pragma unroll
    for (int dt = 0; dt < 8; dt++)
#pragma unroll
      for (int r = 0; r < 8; r++) o[dt][r] *= alpha[r];

    // O += P V   (P: per-wave LDS, same-wave DS ops are in-order; no barrier needed)
    v16bf pf[2];
#pragma unroll
    for (int ks = 0; ks < 2; ks++)
      pf[ks] = frag_a(&sP[wave][ln * 64 + ks * 32], lane);
#pragma unroll
    for (int dt = 0; dt < 8; dt++)
#pragma unroll
      for (int ks = 0; ks < 2; ks++) {
        v16bf vf = frag_b(&sV[cur][(dt * 16 + ln) * 64 + ks * 32], lane);
        o[dt] = wmma_bf16(pf[ks], vf, o[dt]);
      }

#ifdef HAVE_ASYNC_LDS
    wait_async0();        // my prefetch landed
#endif
    __syncthreads();      // all reads of cur done + next block fully in LDS
    cur ^= 1;
  }

#pragma unroll
  for (int dt = 0; dt < 8; dt++)
#pragma unroll
    for (int r = 0; r < 8; r++) {
      int t = qrow0 + r + hl * 8;
      float v = o[dt][r] / lrow[r];
      Yb[((size_t)(b * NT + t)) * NC + h * HDIM + dt * 16 + ln] = f2bf(v);
    }
}

// ---------------- launch ----------------
extern "C" void kernel_launch(void* const* d_in, const int* in_sizes, int n_in,
                              void* d_out, int out_size, void* d_ws, size_t ws_size,
                              hipStream_t stream)
{
  (void)in_sizes; (void)n_in; (void)out_size; (void)ws_size;
  const float* x      = (const float*)d_in[0];
  const float* cosp   = (const float*)d_in[1];
  const float* sinp   = (const float*)d_in[2];
  const float* wq     = (const float*)d_in[3];
  const float* wkv    = (const float*)d_in[4];
  const float* wkup   = (const float*)d_in[5];
  const float* wvup   = (const float*)d_in[6];
  const float* wkrope = (const float*)d_in[7];
  const float* wproj  = (const float*)d_in[8];

  char*  ws  = (char*)d_ws;
  size_t off = 0;
  auto alloc = [&](size_t bytes) -> void* {
    void* p = ws + off;
    off += (bytes + 255) & ~(size_t)255;
    return p;
  };

  const int BT = NB * NT;

  uint16_t* x_bf     = (uint16_t*)alloc((size_t)BT * NC * 2);
  uint16_t* wq_t     = (uint16_t*)alloc((size_t)NC * NC * 2);
  uint16_t* wkv_t    = (uint16_t*)alloc((size_t)512 * NC * 2);
  uint16_t* wkup_t   = (uint16_t*)alloc((size_t)256 * 512 * 2);
  uint16_t* wvup_t   = (uint16_t*)alloc((size_t)512 * 512 * 2);
  uint16_t* wkrope_t = (uint16_t*)alloc((size_t)256 * NC * 2);
  uint16_t* wproj_t  = (uint16_t*)alloc((size_t)NC * NC * 2);
  float*    q_f      = (float*)   alloc((size_t)BT * NC * 4);
  uint16_t* lat_bf   = (uint16_t*)alloc((size_t)BT * 512 * 2);
  float*    knope_f  = (float*)   alloc((size_t)BT * 256 * 4);
  float*    krope_f  = (float*)   alloc((size_t)BT * 256 * 4);
  float*    v_f      = (float*)   alloc((size_t)BT * 512 * 4);
  uint16_t* q_bf     = (uint16_t*)alloc((size_t)NB * NHEAD * NT * HDIM * 2);
  uint16_t* k_bf     = (uint16_t*)alloc((size_t)NB * NKV * NT * HDIM * 2);
  uint16_t* vt_bf    = (uint16_t*)alloc((size_t)NB * NKV * HDIM * NT * 2);
  uint16_t* y_bf     = (uint16_t*)alloc((size_t)BT * NC * 2);

  // converts
  int nx = BT * NC;
  cvt_kernel   <<<(nx + 255) / 256,        256, 0, stream>>>(x,      x_bf, nx);
  cvt_tr_kernel<<<(NC * NC  + 255) / 256,  256, 0, stream>>>(wq,     wq_t,     NC,  NC);
  cvt_tr_kernel<<<(NC * 512 + 255) / 256,  256, 0, stream>>>(wkv,    wkv_t,    NC,  512);
  cvt_tr_kernel<<<(512 * 256 + 255) / 256, 256, 0, stream>>>(wkup,   wkup_t,   512, 256);
  cvt_tr_kernel<<<(512 * 512 + 255) / 256, 256, 0, stream>>>(wvup,   wvup_t,   512, 512);
  cvt_tr_kernel<<<(NC * 256 + 255) / 256,  256, 0, stream>>>(wkrope, wkrope_t, NC,  256);
  cvt_tr_kernel<<<(NC * NC  + 255) / 256,  256, 0, stream>>>(wproj,  wproj_t,  NC,  NC);

  dim3 blk(256);
  // q = x @ wq
  gemm_bf16_kernel<false><<<dim3(NC / 128,  BT / 128), blk, 0, stream>>>(x_bf,   wq_t,     q_f,     BT, NC,  NC);
  // latent (bf16 out) = x @ w_kv
  gemm_bf16_kernel<true> <<<dim3(512 / 128, BT / 128), blk, 0, stream>>>(x_bf,   wkv_t,    lat_bf,  BT, 512, NC);
  // k_nope = latent @ w_kup
  gemm_bf16_kernel<false><<<dim3(256 / 128, BT / 128), blk, 0, stream>>>(lat_bf, wkup_t,   knope_f, BT, 256, 512);
  // k_rope = x @ w_krope
  gemm_bf16_kernel<false><<<dim3(256 / 128, BT / 128), blk, 0, stream>>>(x_bf,   wkrope_t, krope_f, BT, 256, NC);
  // v = latent @ w_vup
  gemm_bf16_kernel<false><<<dim3(512 / 128, BT / 128), blk, 0, stream>>>(lat_bf, wvup_t,   v_f,     BT, 512, 512);

  qprep_kernel<<<(NB * NHEAD * NT) / 8, 256, 0, stream>>>(q_f, cosp, sinp, q_bf);
  kprep_kernel<<<(NB * NKV * NT) / 8,   256, 0, stream>>>(krope_f, knope_f, cosp, sinp, k_bf);
  vtr_kernel  <<<(BT * 512) / 256,      256, 0, stream>>>(v_f, vt_bf);

  fa_kernel<<<dim3(NT / 128, NB * NHEAD), blk, 0, stream>>>(q_bf, k_bf, vt_bf, y_bf);

  // out = y @ w_proj  (f32 straight to d_out)
  gemm_bf16_kernel<false><<<dim3(NC / 128, BT / 128), blk, 0, stream>>>(y_bf, wproj_t, (float*)d_out, BT, NC, NC);
}
